// MSResAttnRefine_39290360823980
// MI455X (gfx1250) — compile-verified, matrix operand order checked by reading, not run on hardware
//
#include <hip/hip_runtime.h>

typedef __attribute__((ext_vector_type(16))) _Float16 v16h;
typedef __attribute__((ext_vector_type(8)))  _Float16 v8h;
typedef __attribute__((ext_vector_type(8)))  float    v8f;

#define BM 128
#define BN 32
#define BK 32
#define LDSTR 72   // halves per LDS row: 144B (16B-aligned b128 reads, low bank conflicts)

// ---------------------------------------------------------------------------
// Generic implicit-GEMM convolution via WMMA f32_16x16x32_f16.
//   M = B*Hout*Wout (spatial), N = Cout, K = Cin*KH*KW
//   mode 0: normal conv, w[Cout][Cin][KH][KW]
//   mode 1: transposed conv (lhs_dilation=2, pad=2, 4x4), w[Cin][Cout][KH][KW], flipped
//   Up to 3 input views emulate channel-concatenated inputs.
//   Each wave computes a 32x16 output tile (2 WMMAs sharing one B fragment).
// ---------------------------------------------------------------------------
struct ConvArgs {
  const float* in0; const float* in1; const float* in2;
  int ct0, co0, cc0;
  int ct1, co1, cc1;
  int ct2, co2, cc2;
  const float* w; const float* bias; float* out;
  int B, Hin, Win, Hout, Wout, Cin, Cout;
  int KH, KW, stride, pad, dil;
  int mode, act, resadd;
  unsigned kmag, wmag;   // ceil(2^20 / (KH*KW)), ceil(2^20 / KW)  (exact for our ranges)
};

__global__ __launch_bounds__(256) void conv_wmma_kernel(ConvArgs a)
{
  __shared__ __align__(16) _Float16 As[BM * LDSTR];
  __shared__ __align__(16) _Float16 Bs[BN * LDSTR];

  const int tid  = threadIdx.x;
  const int HWo  = a.Hout * a.Wout;
  const int HWi  = a.Hin * a.Win;
  const int M    = a.B * HWo;
  const int K    = a.Cin * a.KH * a.KW;
  const int khkw = a.KH * a.KW;
  const int m0   = blockIdx.x * BM;
  const int n0   = blockIdx.y * BN;

  const int wv   = tid >> 5;
  const int lane = tid & 31;
  const int wm   = wv & 3;      // 4 M-subtiles of 32 rows
  const int wn   = wv >> 2;     // 2 N-subtiles of 16 cols
  const int half = lane >> 4;
  const int l    = lane & 15;

  // A cooperative load: 128 rows x 32 cols, 16 halves/thread (2 threads per row)
  const int arow = tid >> 1;
  const int acol = (tid & 1) * 16;
  const int am_g = m0 + arow;
  const bool mvalid = am_g < M;
  int ab = 0, aho = 0, awo = 0;
  if (mvalid) { ab = am_g / HWo; int r = am_g - ab * HWo; aho = r / a.Wout; awo = r - aho * a.Wout; }

  // B cooperative load: 32 n-rows x 32 k, 4 halves/thread (stored transposed [n][k])
  const int brow = tid >> 3;
  const int bcol = (tid & 7) * 4;
  const int bco  = n0 + brow;

  v8f acc0 = {0.f, 0.f, 0.f, 0.f, 0.f, 0.f, 0.f, 0.f};
  v8f acc1 = {0.f, 0.f, 0.f, 0.f, 0.f, 0.f, 0.f, 0.f};

  for (int k0 = 0; k0 < K; k0 += BK) {
    // ---- stage A tile (im2col gather, f32 -> f16) ----
#pragma unroll
    for (int j = 0; j < 16; ++j) {
      int kk = k0 + acol + j;
      float v = 0.f;
      if (mvalid && kk < K) {
        int ci = (int)(((unsigned)kk * a.kmag) >> 20);
        int r  = kk - ci * khkw;
        int kh = (int)(((unsigned)r * a.wmag) >> 20);
        int kw = r - kh * a.KW;
        int ih, iw; bool ok;
        if (a.mode == 0) {
          ih = aho * a.stride - a.pad + kh * a.dil;
          iw = awo * a.stride - a.pad + kw * a.dil;
          ok = ((unsigned)ih < (unsigned)a.Hin) && ((unsigned)iw < (unsigned)a.Win);
        } else {
          int ihu = aho + kh - 2;
          int iwu = awo + kw - 2;
          ok = (ihu >= 0) && (iwu >= 0) && ((ihu & 1) == 0) && ((iwu & 1) == 0);
          ih = ihu >> 1; iw = iwu >> 1;
          ok = ok && (ih < a.Hin) && (iw < a.Win);
        }
        if (ok) {
          if (ci < a.cc0) {
            v = a.in0[((size_t)ab * a.ct0 + a.co0 + ci) * HWi + ih * a.Win + iw];
          } else if (ci < a.cc0 + a.cc1) {
            v = a.in1[((size_t)ab * a.ct1 + a.co1 + (ci - a.cc0)) * HWi + ih * a.Win + iw];
          } else {
            v = a.in2[((size_t)ab * a.ct2 + a.co2 + (ci - a.cc0 - a.cc1)) * HWi + ih * a.Win + iw];
          }
        }
      }
      As[arow * LDSTR + acol + j] = (_Float16)v;
    }
    // ---- stage B tile (weights, transposed [n][k], f32 -> f16) ----
#pragma unroll
    for (int j = 0; j < 4; ++j) {
      int kk = k0 + bcol + j;
      float v = 0.f;
      if (bco < a.Cout && kk < K) {
        if (a.mode == 0) {
          v = a.w[(size_t)bco * K + kk];
          if (j == 0 && kk + BK < K)
            __builtin_prefetch(&a.w[(size_t)bco * K + kk + BK], 0, 0);
        } else {
          int ci = (int)(((unsigned)kk * a.kmag) >> 20);
          int r  = kk - ci * khkw;
          int kh = (int)(((unsigned)r * a.wmag) >> 20);
          int kw = r - kh * a.KW;
          v = a.w[(((size_t)ci * a.Cout + bco) * a.KH + (a.KH - 1 - kh)) * a.KW + (a.KW - 1 - kw)];
        }
      }
      Bs[brow * LDSTR + bcol + j] = (_Float16)v;
    }
    __syncthreads();

    // ---- fragments + 2x WMMA ----
    {
      const int am0 = wm * 32 + l;
      const int am1 = am0 + 16;
      const int bn  = wn * 16 + l;
      // per-lane K layout: lane<16 -> K {0..7,16..23}; lane>=16 -> K {8..15,24..31}
      v8h b0 = *(const v8h*)&Bs[bn * LDSTR + half * 8];
      v8h b1 = *(const v8h*)&Bs[bn * LDSTR + 16 + half * 8];
      v8h x0 = *(const v8h*)&As[am0 * LDSTR + half * 8];
      v8h x1 = *(const v8h*)&As[am0 * LDSTR + 16 + half * 8];
      v8h y0 = *(const v8h*)&As[am1 * LDSTR + half * 8];
      v8h y1 = *(const v8h*)&As[am1 * LDSTR + 16 + half * 8];
      v16h bv, av0, av1;
#pragma unroll
      for (int i = 0; i < 8; ++i) {
        bv[i] = b0[i]; bv[i + 8] = b1[i];
        av0[i] = x0[i]; av0[i + 8] = x1[i];
        av1[i] = y0[i]; av1[i + 8] = y1[i];
      }
      acc0 = __builtin_amdgcn_wmma_f32_16x16x32_f16(false, av0, false, bv, (short)0, acc0, false, false);
      acc1 = __builtin_amdgcn_wmma_f32_16x16x32_f16(false, av1, false, bv, (short)0, acc1, false, false);
    }
    __syncthreads();
  }

  // ---- epilogue: bias + leaky-relu + optional residual add ----
  const int n = n0 + wn * 16 + l;
  if (n < a.Cout) {
    float bb = a.bias ? a.bias[n] : 0.f;
#pragma unroll
    for (int r = 0; r < 8; ++r) {
      int mm = m0 + wm * 32 + half * 8 + r;
      if (mm < M) {
        int ob = mm / HWo;
        int orem = mm - ob * HWo;
        size_t idx = ((size_t)ob * a.Cout + n) * HWo + orem;
        float v = acc0[r] + bb;
        if (a.act) v = (v > 0.f) ? v : 0.2f * v;
        if (a.resadd) v += a.out[idx];
        a.out[idx] = v;
      }
      int mm1 = mm + 16;
      if (mm1 < M) {
        int ob = mm1 / HWo;
        int orem = mm1 - ob * HWo;
        size_t idx = ((size_t)ob * a.Cout + n) * HWo + orem;
        float v = acc1[r] + bb;
        if (a.act) v = (v > 0.f) ? v : 0.2f * v;
        if (a.resadd) v += a.out[idx];
        a.out[idx] = v;
      }
    }
  }
}

// ---------------------------------------------------------------------------
// Elementwise / gather kernels
// ---------------------------------------------------------------------------
__global__ void resize_kernel(const float* in, int ctot, int coff, int C,
                              int B, int Hin, int Win, int Ho, int Wo,
                              float* out, int octot, int ocoff)
{
  int i = blockIdx.x * blockDim.x + threadIdx.x;
  int total = B * C * Ho * Wo;
  if (i >= total) return;
  int wo = i % Wo; int t = i / Wo;
  int ho = t % Ho; t /= Ho;
  int c = t % C; int b = t / C;
  float ys = (Ho > 1) ? (float)ho * (float)(Hin - 1) / (float)(Ho - 1) : 0.f;
  float xs = (Wo > 1) ? (float)wo * (float)(Win - 1) / (float)(Wo - 1) : 0.f;
  int y0 = (int)floorf(ys); if (y0 < 0) y0 = 0; if (y0 > Hin - 2) y0 = Hin - 2;
  int x0 = (int)floorf(xs); if (x0 < 0) x0 = 0; if (x0 > Win - 2) x0 = Win - 2;
  float wy = ys - (float)y0, wx = xs - (float)x0;
  const float* s = in + ((size_t)b * ctot + coff + c) * Hin * Win;
  float v00 = s[y0 * Win + x0];
  float v01 = s[y0 * Win + x0 + 1];
  float v10 = s[(y0 + 1) * Win + x0];
  float v11 = s[(y0 + 1) * Win + x0 + 1];
  float r0 = v00 * (1.f - wy) + v10 * wy;
  float r1 = v01 * (1.f - wy) + v11 * wy;
  out[((size_t)b * octot + ocoff + c) * Ho * Wo + ho * Wo + wo] = r0 * (1.f - wx) + r1 * wx;
}

// (B,Hin,Win,K) -> (B,Ho,Wo,K) bilinear (for prob-map upsampling)
__global__ void resize_chlast_kernel(const float* in, int B, int Hin, int Win, int K,
                                     int Ho, int Wo, float* out)
{
  int i = blockIdx.x * blockDim.x + threadIdx.x;
  int total = B * Ho * Wo * K;
  if (i >= total) return;
  int k = i % K; int t = i / K;
  int wo = t % Wo; t /= Wo;
  int ho = t % Ho; int b = t / Ho;
  float ys = (Ho > 1) ? (float)ho * (float)(Hin - 1) / (float)(Ho - 1) : 0.f;
  float xs = (Wo > 1) ? (float)wo * (float)(Win - 1) / (float)(Wo - 1) : 0.f;
  int y0 = (int)floorf(ys); if (y0 < 0) y0 = 0; if (y0 > Hin - 2) y0 = Hin - 2;
  int x0 = (int)floorf(xs); if (x0 < 0) x0 = 0; if (x0 > Win - 2) x0 = Win - 2;
  float wy = ys - (float)y0, wx = xs - (float)x0;
  const float* s = in + (size_t)b * Hin * Win * K;
  float v00 = s[((size_t)y0 * Win + x0) * K + k];
  float v01 = s[((size_t)y0 * Win + x0 + 1) * K + k];
  float v10 = s[((size_t)(y0 + 1) * Win + x0) * K + k];
  float v11 = s[((size_t)(y0 + 1) * Win + x0 + 1) * K + k];
  float r0 = v00 * (1.f - wy) + v10 * wy;
  float r1 = v01 * (1.f - wy) + v11 * wy;
  out[(((size_t)b * Ho + ho) * Wo + wo) * K + k] = r0 * (1.f - wx) + r1 * wx;
}

__global__ void rnorm_kernel(const float* x, int B, int C, int HW, float* rn)
{
  int i = blockIdx.x * blockDim.x + threadIdx.x;
  if (i >= B * HW) return;
  int b = i / HW, p = i - b * HW;
  const float* s = x + (size_t)b * C * HW + p;
  float acc = 0.f;
  for (int c = 0; c < C; ++c) { float v = s[(size_t)c * HW]; acc += v * v; }
  rn[i] = rsqrtf(acc);
}

// sim[(b,h,w), k] for k in [0,90): k<45 -> forward, else backward; dy=k%45/9-2, dx=k%45%9-4
__global__ void corr_kernel(const float* x, const float* f, const float* g,
                            const float* rnx, const float* rnf, const float* rng,
                            float* sim, int B, int C, int H, int W)
{
  int i = blockIdx.x * blockDim.x + threadIdx.x;
  int total = B * H * W * 90;
  if (i >= total) return;
  int k = i % 90; int p = i / 90;
  int w = p % W; int t = p / W;
  int h = t % H; int b = t / H;
  int tt = k / 45; int kk = k - tt * 45;
  int dy = kk / 9 - 2, dx = kk % 9 - 4;
  int hh = h + dy, ww = w + dx;
  float sv = 0.f;
  if ((unsigned)hh < (unsigned)H && (unsigned)ww < (unsigned)W) {
    const float* tp = tt ? g : f;
    const float* rt = tt ? rng : rnf;
    const float* xs = x  + (size_t)b * C * H * W + h * W + w;
    const float* ts = tp + (size_t)b * C * H * W + hh * W + ww;
    float acc = 0.f;
    for (int c = 0; c < C; ++c) acc += xs[(size_t)c * H * W] * ts[(size_t)c * H * W];
    sv = acc * rnx[(b * H + h) * W + w] * rt[(b * H + hh) * W + ww];
  }
  sim[(size_t)p * 90 + k] = sv;
}

__global__ void softmax90_kernel(float* prob, int N)
{
  int i = blockIdx.x * blockDim.x + threadIdx.x;
  if (i >= N) return;
  float* p = prob + (size_t)i * 90;
  float m = p[0];
  for (int j = 1; j < 90; ++j) m = fmaxf(m, p[j]);
  float s = 0.f;
  for (int j = 0; j < 90; ++j) { float e = expf(p[j] - m); p[j] = e; s += e; }
  float inv = 1.f / s;
  for (int j = 0; j < 90; ++j) p[j] *= inv;
}

__global__ void wlow_kernel(const float* feat, const float* prob, int poff,
                            float* out, int B, int C, int H, int W)
{
  int i = blockIdx.x * blockDim.x + threadIdx.x;
  int total = B * C * H * W;
  if (i >= total) return;
  int w = i % W; int t = i / W;
  int h = t % H; t /= H;
  int c = t % C; int b = t / C;
  const float* pp = prob + ((size_t)(b * H + h) * W + w) * 90 + poff;
  const float* fs = feat + ((size_t)b * C + c) * H * W;
  float num = 0.f, den = 0.f;
  for (int k = 0; k < 45; ++k) {
    float pk = pp[k]; den += pk;
    int hh = h + k / 9 - 2, ww = w + (k % 9) - 4;
    if ((unsigned)hh < (unsigned)H && (unsigned)ww < (unsigned)W)
      num += pk * fs[hh * W + ww];
  }
  out[i] = num / den;
}

__global__ void wfull_kernel(const float* f, const float* g, const float* prob,
                             float* out, int B, int C, int H, int W)
{
  int i = blockIdx.x * blockDim.x + threadIdx.x;
  int total = B * C * H * W;
  if (i >= total) return;
  int w = i % W; int t = i / W;
  int h = t % H; t /= H;
  int c = t % C; int b = t / C;
  const float* pp = prob + ((size_t)(b * H + h) * W + w) * 90;
  const float* fs = f + ((size_t)b * C + c) * H * W;
  const float* gs = g + ((size_t)b * C + c) * H * W;
  float acc = 0.f;
  for (int k = 0; k < 45; ++k) {
    int hh = h + k / 9 - 2, ww = w + (k % 9) - 4;
    if ((unsigned)hh < (unsigned)H && (unsigned)ww < (unsigned)W) {
      acc += pp[k] * fs[hh * W + ww];
      acc += pp[45 + k] * gs[hh * W + ww];
    }
  }
  out[i] = acc;
}

__global__ void add_kernel(float* out, const float* a, int n)
{
  int i = blockIdx.x * blockDim.x + threadIdx.x;
  if (i < n) out[i] += a[i];
}

// ---------------------------------------------------------------------------
// Host orchestration
// ---------------------------------------------------------------------------
struct View { const float* p; int ctot, coff, ccnt; };
static inline View NV() { return View{nullptr, 0, 0, 0}; }
static inline View FV(const float* p, int c) { return View{p, c, 0, c}; }

static void conv_go(hipStream_t st, View v0, View v1, View v2,
                    const float* w, const float* bias, float* out,
                    int B, int Hin, int Win, int Hout, int Wout, int Cout,
                    int KH, int KW, int stride, int pad, int dil,
                    int mode, int act, int resadd)
{
  ConvArgs a;
  a.in0 = v0.p; a.ct0 = v0.ctot; a.co0 = v0.coff; a.cc0 = v0.ccnt;
  a.in1 = v1.p; a.ct1 = v1.ctot; a.co1 = v1.coff; a.cc1 = v1.ccnt;
  a.in2 = v2.p; a.ct2 = v2.ctot; a.co2 = v2.coff; a.cc2 = v2.ccnt;
  a.w = w; a.bias = bias; a.out = out;
  a.B = B; a.Hin = Hin; a.Win = Win; a.Hout = Hout; a.Wout = Wout;
  a.Cin = v0.ccnt + v1.ccnt + v2.ccnt; a.Cout = Cout;
  a.KH = KH; a.KW = KW; a.stride = stride; a.pad = pad; a.dil = dil;
  a.mode = mode; a.act = act; a.resadd = resadd;
  int khkw = KH * KW;
  a.kmag = (unsigned)(((1u << 20) + khkw - 1) / khkw);
  a.wmag = (unsigned)(((1u << 20) + KW - 1) / KW);
  int M = B * Hout * Wout;
  dim3 grid((unsigned)((M + BM - 1) / BM), (unsigned)((Cout + BN - 1) / BN));
  conv_wmma_kernel<<<grid, dim3(256), 0, st>>>(a);
}

// Param leaf indices (jax tree: dict keys sorted, lists/tuples in order)
enum {
  P_AE1 = 4, P_AE2 = 8, P_AFUSE = 12, P_AIMGF = 16, P_AIN = 20,
  P_DEC1 = 24, P_DEC2 = 30, P_ATR = 36,
  P_IE1 = 44, P_IE2 = 48, P_IFUSE = 52, P_IIN = 56, P_IN = 60, P_OUT = 64
};

extern "C" void kernel_launch(void* const* d_in, const int* in_sizes, int n_in,
                              void* d_out, int out_size, void* d_ws, size_t ws_size,
                              hipStream_t stream)
{
  (void)in_sizes; (void)n_in; (void)out_size; (void)ws_size;
  const float* cimg = (const float*)d_in[0];
  const float* cseg = (const float*)d_in[1];
  const float* nimg = (const float*)d_in[2];
  const float* nseg = (const float*)d_in[3];
  auto P = [&](int i) { return (const float*)d_in[i]; };

  const int B = 2, H0 = 256, W0 = 512;
  const size_t U = (size_t)B * H0 * W0;        // elements per full-res channel-map

  float* ws = (float*)d_ws;
  size_t off = 0;
  auto alloc = [&](size_t n) { float* p = ws + off; off += n; return p; };

  float* PROB1 = alloc((size_t)B * 32 * 64 * 90);   // persists si=1 -> si=0
  float* XC = alloc(23 * U / 4);
  float* FC = alloc(23 * U / 4);
  float* BC = alloc(23 * U / 4);
  float* XIN = alloc(64 * U);
  float* FIN = alloc(64 * U);
  float* BIN = alloc(64 * U);
  float* T32 = alloc(32 * U);
  float* T64A = alloc(64 * U);
  float* T64B = alloc(64 * U);
  float* T64H1 = alloc(64 * U / 4);
  float* T64H2 = alloc(64 * U / 4);
  float* T128Q = alloc(128 * U / 16);
  float* XE2 = alloc(128 * U / 16);
  float* FE2 = alloc(128 * U / 16);
  float* BE2 = alloc(128 * U / 16);
  float* FWB = alloc(128 * U / 16);
  float* BWB = alloc(128 * U / 16);
  float* NBW = alloc(128 * U / 16);
  float* AF  = alloc(128 * U / 16);
  float* IE2 = alloc(128 * U / 16);
  float* ATA = alloc(128 * U / 16);
  float* ATB = alloc(128 * U / 16);
  float* FUSEDQ = alloc(128 * U / 16);
  float* RNX = alloc(U / 16);
  float* RNF = alloc(U / 16);
  float* RNB = alloc(U / 16);
  float* PROB0 = alloc((size_t)B * 64 * 128 * 90);
  float* PMUP  = alloc((size_t)B * 64 * 128 * 90);
  float* D2 = alloc(64 * U / 4);
  float* D1 = XIN;   // XIN dead before decoder stage

  // single-view 3x3 conv helper
  auto c3 = [&](const float* in, int Cin, int wi, float* outp, int Cout,
                int Hin, int Win, int stride, int pad, int dil, int act, int res) {
    int Ho = (stride == 2) ? Hin / 2 : Hin;
    int Wo = (stride == 2) ? Win / 2 : Win;
    conv_go(stream, FV(in, Cin), NV(), NV(), P(wi), P(wi + 1), outp,
            B, Hin, Win, Ho, Wo, Cout, 3, 3, stride, pad, dil, 0, act, res);
  };
  // transposed 4x4 conv (stride 2 upsample), lrelu
  auto ctr = [&](const float* in, int Cin, int wi, float* outp, int Cout, int Hin, int Win) {
    conv_go(stream, FV(in, Cin), NV(), NV(), P(wi), P(wi + 1), outp,
            B, Hin, Win, Hin * 2, Win * 2, Cout, 4, 4, 1, 0, 1, 1, 1, 0);
  };
  // shared encoder: input -> *_in (64ch, kept) and *_e2 (128ch @ H/4)
  auto enc = [&](View v0, View v1, float* dst_in, float* dst_e2, int H, int W) {
    int H2 = H / 2, W2 = W / 2, H4 = H / 4, W4 = W / 4;
    conv_go(stream, v0, v1, NV(), P(P_IN + 0), P(P_IN + 1), T32,
            B, H, W, H, W, 32, 3, 3, 1, 1, 1, 0, 1, 0);
    c3(T32, 32, P_IN + 2, dst_in, 64, H, W, 1, 1, 1, 1, 0);
    c3(dst_in, 64, P_AIN + 0, T64A, 64, H, W, 1, 1, 1, 1, 0);
    c3(T64A, 64, P_AIN + 2, T64B, 64, H, W, 1, 1, 1, 1, 0);
    c3(T64B, 64, P_AE1 + 0, T64H1, 64, H, W, 2, 1, 1, 1, 0);
    c3(T64H1, 64, P_AE1 + 2, T64H2, 64, H2, W2, 1, 1, 1, 1, 0);
    c3(T64H2, 64, P_AE2 + 0, T128Q, 128, H2, W2, 2, 1, 1, 1, 0);
    c3(T128Q, 128, P_AE2 + 2, dst_e2, 128, H4, W4, 1, 1, 1, 1, 0);
    (void)H4; (void)W4;
  };

  for (int si = 1; si >= 0; --si) {
    const int H = H0 >> si, W = W0 >> si;
    const int H2 = H / 2, W2 = W / 2, H4 = H / 4, W4 = W / 4;

    View vx0, vx1, vf0, vf1, vb0, vb1;
    if (si == 1) {
      int n;
      n = B * 3 * H * W;
      resize_kernel<<<(n + 255) / 256, 256, 0, stream>>>(cimg, 3, 0, 3, B, H0, W0, H, W, XC, 23, 0);
      n = B * 20 * H * W;
      resize_kernel<<<(n + 255) / 256, 256, 0, stream>>>(cseg, 20, 0, 20, B, H0, W0, H, W, XC, 23, 3);
      n = B * 3 * H * W;
      resize_kernel<<<(n + 255) / 256, 256, 0, stream>>>(nimg, 6, 0, 3, B, H0, W0, H, W, FC, 23, 0);
      n = B * 20 * H * W;
      resize_kernel<<<(n + 255) / 256, 256, 0, stream>>>(nseg, 40, 0, 20, B, H0, W0, H, W, FC, 23, 3);
      n = B * 3 * H * W;
      resize_kernel<<<(n + 255) / 256, 256, 0, stream>>>(nimg, 6, 3, 3, B, H0, W0, H, W, BC, 23, 0);
      n = B * 20 * H * W;
      resize_kernel<<<(n + 255) / 256, 256, 0, stream>>>(nseg, 40, 20, 20, B, H0, W0, H, W, BC, 23, 3);
      vx0 = FV(XC, 23); vx1 = NV();
      vf0 = FV(FC, 23); vf1 = NV();
      vb0 = FV(BC, 23); vb1 = NV();
    } else {
      vx0 = View{cimg, 3, 0, 3};   vx1 = View{cseg, 20, 0, 20};
      vf0 = View{nimg, 6, 0, 3};   vf1 = View{nseg, 40, 0, 20};
      vb0 = View{nimg, 6, 3, 3};   vb1 = View{nseg, 40, 20, 20};
    }

    enc(vx0, vx1, XIN, XE2, H, W);
    enc(vf0, vf1, FIN, FE2, H, W);
    enc(vb0, vb1, BIN, BE2, H, W);

    float* fw; float* bw;
    if (si == 1) {
      fw = FE2; bw = BE2;
    } else {
      // upsample prob map from si=1 (B,32,64,90) -> (B,64,128,90)
      int n = B * H4 * W4 * 90;
      resize_chlast_kernel<<<(n + 255) / 256, 256, 0, stream>>>(PROB1, B, 32, 64, 90, H4, W4, PMUP);
      int nq = B * 128 * H4 * W4;
      wlow_kernel<<<(nq + 255) / 256, 256, 0, stream>>>(FE2, PMUP, 0,  FWB, B, 128, H4, W4);
      wlow_kernel<<<(nq + 255) / 256, 256, 0, stream>>>(BE2, PMUP, 45, BWB, B, 128, H4, W4);
      c3(FWB, 128, P_AFUSE + 0, T128Q, 128, H4, W4, 1, 1, 1, 1, 0);
      c3(T128Q, 128, P_AFUSE + 2, FWB, 128, H4, W4, 1, 1, 1, 1, 0);
      c3(BWB, 128, P_AFUSE + 0, T128Q, 128, H4, W4, 1, 1, 1, 1, 0);
      c3(T128Q, 128, P_AFUSE + 2, BWB, 128, H4, W4, 1, 1, 1, 1, 0);
      fw = FWB; bw = BWB;
    }

    // corrmap: normalized correlations over 5x9 window for both neighbors + softmax
    int np = B * H4 * W4;
    rnorm_kernel<<<(np + 255) / 256, 256, 0, stream>>>(XE2, B, 128, H4 * W4, RNX);
    rnorm_kernel<<<(np + 255) / 256, 256, 0, stream>>>(fw,  B, 128, H4 * W4, RNF);
    rnorm_kernel<<<(np + 255) / 256, 256, 0, stream>>>(bw,  B, 128, H4 * W4, RNB);
    float* prob = (si == 1) ? PROB1 : PROB0;
    int nc = np * 90;
    corr_kernel<<<(nc + 255) / 256, 256, 0, stream>>>(XE2, fw, bw, RNX, RNF, RNB, prob, B, 128, H4, W4);
    softmax90_kernel<<<(np + 255) / 256, 256, 0, stream>>>(prob, np);

    if (si == 0) {
      // everything below only affects the final output at the finest scale
      int nq = B * 128 * H4 * W4;
      wfull_kernel<<<(nq + 255) / 256, 256, 0, stream>>>(FE2, BE2, prob, NBW, B, 128, H4, W4);
      // attn_img_fuse on concat(x_e2, nbw) : 256 -> 128 -> 128
      conv_go(stream, FV(XE2, 128), FV(NBW, 128), NV(), P(P_AIMGF + 0), P(P_AIMGF + 1), T128Q,
              B, H4, W4, H4, W4, 128, 3, 3, 1, 1, 1, 0, 1, 0);
      c3(T128Q, 128, P_AIMGF + 2, AF, 128, H4, W4, 1, 1, 1, 1, 0);
      // img_input on concat(x_in, f_in, b_in) : 192 -> 64 -> 64
      conv_go(stream, FV(XIN, 64), FV(FIN, 64), FV(BIN, 64), P(P_IIN + 0), P(P_IIN + 1), T64A,
              B, H, W, H, W, 64, 3, 3, 1, 1, 1, 0, 1, 0);
      c3(T64A, 64, P_IIN + 2, T64B, 64, H, W, 1, 1, 1, 1, 0);          // img_in
      // img encoders
      c3(T64B, 64, P_IE1 + 0, T64H1, 64, H, W, 2, 1, 1, 1, 0);
      c3(T64H1, 64, P_IE1 + 2, T64H2, 64, H2, W2, 1, 1, 1, 1, 0);      // ie1
      c3(T64H2, 64, P_IE2 + 0, T128Q, 128, H2, W2, 2, 1, 1, 1, 0);
      c3(T128Q, 128, P_IE2 + 2, IE2, 128, H4, W4, 1, 1, 1, 1, 0);      // ie2
      // atrous chain d = 1,2,4,8
      c3(IE2, 128, P_ATR + 0, ATA, 128, H4, W4, 1, 1, 1, 1, 0);
      c3(ATA, 128, P_ATR + 2, ATB, 128, H4, W4, 1, 2, 2, 1, 0);
      c3(ATB, 128, P_ATR + 4, ATA, 128, H4, W4, 1, 4, 4, 1, 0);
      c3(ATA, 128, P_ATR + 6, ATB, 128, H4, W4, 1, 8, 8, 1, 0);
      // img_fuse on concat(at, attn_fused) : 256 -> 128 -> 128
      conv_go(stream, FV(ATB, 128), FV(AF, 128), NV(), P(P_IFUSE + 0), P(P_IFUSE + 1), T128Q,
              B, H4, W4, H4, W4, 128, 3, 3, 1, 1, 1, 0, 1, 0);
      c3(T128Q, 128, P_IFUSE + 2, FUSEDQ, 128, H4, W4, 1, 1, 1, 1, 0);
      // decoder level 2
      ctr(FUSEDQ, 128, P_DEC2 + 0, D2, 64, H4, W4);                    // lrelu(conv_t)
      c3(D2, 64, P_DEC2 + 2, T64H1, 64, H2, W2, 1, 1, 1, 1, 0);
      c3(T64H1, 64, P_DEC2 + 4, D2, 64, H2, W2, 1, 1, 1, 0, 1);        // residual add
      // decoder level 1
      ctr(D2, 64, P_DEC1 + 0, D1, 64, H2, W2);
      c3(D1, 64, P_DEC1 + 2, T64A, 64, H, W, 1, 1, 1, 1, 0);
      c3(T64A, 64, P_DEC1 + 4, D1, 64, H, W, 1, 1, 1, 0, 1);           // residual add
      // output head
      c3(D1, 64, P_OUT + 0, T64B, 64, H, W, 1, 1, 1, 1, 0);
      c3(T64B, 64, P_OUT + 2, T32, 32, H, W, 1, 1, 1, 1, 0);
      c3(T32, 32, P_OUT + 4, (float*)d_out, 3, H, W, 1, 1, 1, 0, 0);
      int n = B * 3 * H * W;
      add_kernel<<<(n + 255) / 256, 256, 0, stream>>>((float*)d_out, cimg, n);
    }
  }
}
